// LearnedSimulator_20083267076600
// MI455X (gfx1250) — matrix operand order repaired and checked by reading, not compiled, not run on hardware
//
#include <hip/hip_runtime.h>
#include <cstdint>
#include <cstddef>

// ---------------------------------------------------------------------------
// LearnedSimulator (Encode-Process-Decode GNN) for gfx1250 / MI455X.
// All GEMMs run on V_WMMA_F32_16X16X32_BF16 (wave32). Activations are stored
// bf16 to halve HBM traffic (memory-bound workload, ~7 GB total @ 23.3 TB/s).
// Each wave computes a 16-row tile through a fused 3-layer MLP + LayerNorm +
// residual (+ scatter-add for the edge block) without spilling intermediates.
// ---------------------------------------------------------------------------

#define DEVFN __device__ __forceinline__

constexpr int kNodes  = 20000;
constexpr int kEdges  = 320000;
constexpr int kNodeIn = 37;
constexpr int kEdgeIn = 4;
constexpr int kLatent = 128;
constexpr int kSteps  = 10;

typedef __bf16 bf16_t;
typedef bf16_t v16bf __attribute__((ext_vector_type(16)));
typedef bf16_t v8bf  __attribute__((ext_vector_type(8)));
typedef float  v8f   __attribute__((ext_vector_type(8)));

DEVFN v16bf comb(v8bf lo, v8bf hi) {
  return __builtin_shufflevector(lo, hi, 0,1,2,3,4,5,6,7,8,9,10,11,12,13,14,15);
}

DEVFN v8f wmma_bf16(v16bf a, v16bf b, v8f c) {
  return __builtin_amdgcn_wmma_f32_16x16x32_bf16(false, a, false, b, (short)0, c,
                                                 false, false);
}

// ---------------------------------------------------------------------------
// Weight packing: W[K,N] row-major f32 -> per-fragment per-lane bf16 layout.
// B fragment (K=32 tile, N=16 tile): lane<16 holds n=lane, k=kt*32+0..15;
// lane>=16 holds n=lane-16, k=kt*32+16..31. 16 contiguous bf16 per lane.
// ---------------------------------------------------------------------------
struct PackDesc { const float* src; bf16_t* dst; int Kact, KT, Nact, NT; };
struct PackArgs { PackDesc d[69]; };

__global__ void pack_weights_kernel(PackArgs P) {
  const PackDesc D = P.d[blockIdx.y];
  const int total = D.KT * D.NT * 512;
  for (int idx = blockIdx.x * blockDim.x + threadIdx.x; idx < total;
       idx += gridDim.x * blockDim.x) {
    int e  = idx & 15;
    int ln = (idx >> 4) & 31;
    int f  = idx >> 9;
    int nt = f % D.NT;
    int kt = f / D.NT;
    int k  = kt * 32 + ((ln & 16) ? 16 : 0) + e;
    int n  = nt * 16 + (ln & 15);
    float v = (k < D.Kact && n < D.Nact) ? D.src[(size_t)k * D.Nact + n] : 0.0f;
    D.dst[idx] = (bf16_t)v;
  }
}

// ---------------------------------------------------------------------------
// Fused 3-layer MLP kernel (one 16-row tile per wave, 4 waves / block).
// ---------------------------------------------------------------------------
enum { M_ENC_NODE = 0, M_ENC_EDGE = 1, M_PROC_EDGE = 2, M_PROC_NODE = 3, M_DEC = 4 };

struct MlpArgs {
  const float*  a_f32;    // raw f32 input (encoders)
  const bf16_t* e_in;     // edge latents (proc edge)
  const bf16_t* x_in;     // node latents (proc edge/node, dec)
  const float*  agg_in;   // aggregated edge messages (proc node)
  const int*    recv;
  const int*    send;
  const bf16_t *W0, *W1, *W2;     // packed bf16 fragment weights
  const float  *b0, *b1, *b2;     // f32 biases (read direct from inputs)
  const bf16_t* res;      // residual source (x_enc / e_enc)
  bf16_t* out_enc;        // encoder output buffer
  bf16_t* out_cur;        // running latent buffer
  float*  agg_rmw;        // scatter target (edge) / zeroed after use (node)
  float*  dec_out;        // final [N,3] output
  int rows, nb2;
};

template <int KT, int NT, typename F>
DEVFN void gemm(F loadA, const bf16_t* W, const float* bias, int nbias, int lane,
                v8f (&acc)[NT]) {
  const int col = lane & 15;
#pragma unroll
  for (int nt = 0; nt < NT; ++nt) {
    int bc = nt * 16 + col;
    float bv = (bc < nbias) ? bias[bc] : 0.0f;
#pragma unroll
    for (int i = 0; i < 8; ++i) acc[nt][i] = bv;
  }
#pragma unroll
  for (int kt = 0; kt < KT; ++kt) {
    v16bf a = loadA(kt);
    if (kt + 1 < KT)  // warm L2/WGP$ for the next K-tile's weight fragments
      __builtin_prefetch(W + ((((size_t)(kt + 1) * NT) * 32 + lane) << 4), 0, 1);
#pragma unroll
    for (int nt = 0; nt < NT; ++nt) {
      const bf16_t* q = W + ((((size_t)kt * NT + nt) * 32 + lane) << 4);
      v16bf b = comb(*(const v8bf*)q, *(const v8bf*)(q + 8));
      acc[nt] = wmma_bf16(a, b, acc[nt]);
    }
  }
}

DEVFN void relu8(v8f (&acc)[8]) {
#pragma unroll
  for (int nt = 0; nt < 8; ++nt)
#pragma unroll
    for (int i = 0; i < 8; ++i) acc[nt][i] = fmaxf(acc[nt][i], 0.0f);
}

// C/D layout: VGPR i, lanes 0-15 -> (row=i, col=lane); lanes 16-31 -> (row=i+8).
DEVFN void store_tile8(bf16_t* t, v8f (&acc)[8], int lane) {
  const int rb = (lane & 16) ? 8 : 0, col = lane & 15;
#pragma unroll
  for (int nt = 0; nt < 8; ++nt)
#pragma unroll
    for (int i = 0; i < 8; ++i)
      t[(rb + i) * kLatent + nt * 16 + col] = (bf16_t)acc[nt][i];
}

DEVFN void layernorm8(v8f (&acc)[8]) {
#pragma unroll
  for (int i = 0; i < 8; ++i) {
    float s = 0.0f, q = 0.0f;
#pragma unroll
    for (int nt = 0; nt < 8; ++nt) { float v = acc[nt][i]; s += v; q += v * v; }
#pragma unroll
    for (int m = 1; m < 16; m <<= 1) {  // reduce within the 16-lane half-wave
      s += __shfl_xor(s, m, 16);
      q += __shfl_xor(q, m, 16);
    }
    float mean = s * (1.0f / 128.0f);
    float var  = q * (1.0f / 128.0f) - mean * mean;
    float inv  = rsqrtf(var + 1e-5f);
#pragma unroll
    for (int nt = 0; nt < 8; ++nt) acc[nt][i] = (acc[nt][i] - mean) * inv;
  }
}

template <int MODE, int KT1>
__global__ __launch_bounds__(128) void mlp3_kernel(MlpArgs A) {
  // Compile-time layer-1 input width for the encoders.
  constexpr int ACOLS = (MODE == M_ENC_NODE) ? kNodeIn
                        : (MODE == M_ENC_EDGE) ? kEdgeIn : 0;

  const int lane = threadIdx.x & 31;
  const int wid  = threadIdx.x >> 5;
  const int ntiles = A.rows >> 4;
  const int tile = blockIdx.x * 4 + wid;
  if (tile >= ntiles) return;  // whole-wave exit; EXEC stays all-ones for WMMA
  const int row0 = tile << 4;

  __shared__ __align__(16) bf16_t smem[4][16 * kLatent];
  bf16_t* t = smem[wid];

  const int r = row0 + (lane & 15);
  const float*  af      = nullptr;
  const bf16_t* row_e   = nullptr;
  const bf16_t* row_xr  = nullptr;
  const bf16_t* row_xs  = nullptr;
  const bf16_t* row_x   = nullptr;
  const float*  row_agg = nullptr;
  if (MODE == M_ENC_NODE || MODE == M_ENC_EDGE) af = A.a_f32 + (size_t)r * ACOLS;
  if (MODE == M_PROC_EDGE) {
    row_e  = A.e_in + (size_t)r * kLatent;
    row_xr = A.x_in + (size_t)A.recv[r] * kLatent;
    row_xs = A.x_in + (size_t)A.send[r] * kLatent;
  }
  if (MODE == M_PROC_NODE) {
    row_x   = A.x_in + (size_t)r * kLatent;
    row_agg = A.agg_in + (size_t)r * kLatent;
  }
  if (MODE == M_DEC) row_x = A.x_in + (size_t)r * kLatent;

  // 8-element chunk of layer-1 input at column k (chunks never straddle a
  // 128-wide concat segment, so each chunk is one native 16B load).
  auto chunkA = [&](int k) -> v8bf {
    if (MODE == M_ENC_NODE || MODE == M_ENC_EDGE) {
      // The k < ACOLS test is per-lane divergent (lanes 16..31 carry a +16 k
      // offset). Load unconditionally from a clamped in-bounds index and
      // select zero afterwards: lowers to v_cndmask, no exec-mask branches.
      v8bf v;
#pragma unroll
      for (int e = 0; e < 8; ++e) {
        bool ok = (k + e) < ACOLS;
        int kk = ok ? (k + e) : 0;
        float val = af[kk];
        v[e] = ok ? (bf16_t)val : (bf16_t)0.0f;
      }
      return v;
    } else if (MODE == M_PROC_EDGE) {
      const bf16_t* p = (k < kLatent)     ? row_e + k
                        : (k < 2*kLatent) ? row_xr + (k - kLatent)
                                          : row_xs + (k - 2*kLatent);
      return *(const v8bf*)p;
    } else if (MODE == M_PROC_NODE) {
      if (k < kLatent) return *(const v8bf*)(row_x + k);
      v8bf v; const float* p = row_agg + (k - kLatent);
#pragma unroll
      for (int e = 0; e < 8; ++e) v[e] = (bf16_t)p[e];
      return v;
    } else {
      return *(const v8bf*)(row_x + k);
    }
  };
  // A fragment in this code's contiguous k order (same as the B packing): lane holds row=lane&15, k-chunks at
  // kt*32 + (lane>=16 ? 16 : 0) and +8.
  auto loadA1 = [&](int kt) -> v16bf {
    int kl = kt * 32 + ((lane & 16) ? 16 : 0);
    return comb(chunkA(kl), chunkA(kl + 8));
  };
  auto loadA_lds = [&](int kt) -> v16bf {
    int rr = lane & 15;
    int kl = kt * 32 + ((lane & 16) ? 16 : 0);
    return comb(*(const v8bf*)(t + rr * kLatent + kl),
                *(const v8bf*)(t + rr * kLatent + kl + 8));
  };

  v8f acc[8];
  gemm<KT1, 8>(loadA1, A.W0, A.b0, 128, lane, acc);
  relu8(acc); store_tile8(t, acc, lane);
  gemm<4, 8>(loadA_lds, A.W1, A.b1, 128, lane, acc);
  relu8(acc); store_tile8(t, acc, lane);

  if (MODE == M_DEC) {  // final linear, N padded 3->16, no LayerNorm
    v8f accd[1];
    gemm<4, 1>(loadA_lds, A.W2, A.b2, A.nb2, lane, accd);
    int col = lane & 15;
    if (col < 3) {
      int rb = (lane & 16) ? 8 : 0;
#pragma unroll
      for (int i = 0; i < 8; ++i)
        A.dec_out[(size_t)(row0 + rb + i) * 3 + col] = accd[0][i];
    }
    return;
  }

  gemm<4, 8>(loadA_lds, A.W2, A.b2, 128, lane, acc);
  layernorm8(acc);
  store_tile8(t, acc, lane);  // LN output staged in LDS for coalesced epilogue

#pragma unroll
  for (int it = 0; it < 8; ++it) {  // 256 chunks of 8 bf16 over the 16x128 tile
    int c  = it * 32 + lane;
    int rr = c >> 4;
    int cc = (c & 15) * 8;
    v8bf v = *(const v8bf*)(t + rr * kLatent + cc);
    size_t off = (size_t)(row0 + rr) * kLatent + cc;
    if (MODE == M_ENC_NODE || MODE == M_ENC_EDGE) {
      *(v8bf*)(A.out_enc + off) = v;
      *(v8bf*)(A.out_cur + off) = v;
    } else {
      v8bf e0 = *(const v8bf*)(A.res + off);
      v8bf nv;
#pragma unroll
      for (int e = 0; e < 8; ++e) nv[e] = (bf16_t)((float)v[e] + (float)e0[e]);
      *(v8bf*)(A.out_cur + off) = nv;
      if (MODE == M_PROC_EDGE) {  // segment_sum: f32 atomics, agg is L2-resident
        int dst = A.recv[row0 + rr];
        float* ap = A.agg_rmw + (size_t)dst * kLatent + cc;
#pragma unroll
        for (int e = 0; e < 8; ++e) atomicAdd(ap + e, (float)v[e]);
      }
    }
  }
  if (MODE == M_PROC_NODE) {  // consumed agg for this tile: zero it for next step
#pragma unroll
    for (int it = 0; it < 16; ++it) {
      int idx = it * 32 + lane;
      int rr  = idx >> 5;
      int cc  = (idx & 31) * 4;
      *(float4*)(A.agg_rmw + (size_t)(row0 + rr) * kLatent + cc) =
          make_float4(0.f, 0.f, 0.f, 0.f);
    }
  }
}

// ---------------------------------------------------------------------------
// Host side
// ---------------------------------------------------------------------------
extern "C" void kernel_launch(void* const* d_in, const int* in_sizes, int n_in,
                              void* d_out, int out_size, void* d_ws, size_t ws_size,
                              hipStream_t stream) {
  (void)n_in; (void)out_size; (void)ws_size;
  const float *x, *eattr;
  const int *recv, *send;
  const float *encn[6], *ence[6], *pe[6], *pn[6], *dec[6];
  // Detect pytree flattening order: insertion (x first) vs alphabetical.
  if (in_sizes[0] == kNodes * kNodeIn) {
    x = (const float*)d_in[0]; eattr = (const float*)d_in[1];
    for (int i = 0; i < 6; ++i) {
      encn[i] = (const float*)d_in[2 + i];  ence[i] = (const float*)d_in[8 + i];
      pe[i]   = (const float*)d_in[14 + i]; pn[i]   = (const float*)d_in[20 + i];
      dec[i]  = (const float*)d_in[26 + i];
    }
    recv = (const int*)d_in[32]; send = (const int*)d_in[33];
  } else {
    eattr = (const float*)d_in[0];
    for (int i = 0; i < 6; ++i) {
      dec[i]  = (const float*)d_in[1 + i];  ence[i] = (const float*)d_in[7 + i];
      encn[i] = (const float*)d_in[13 + i]; pe[i]   = (const float*)d_in[19 + i];
      pn[i]   = (const float*)d_in[25 + i];
    }
    recv = (const int*)d_in[31]; send = (const int*)d_in[32];
    x = (const float*)d_in[33];
  }

  // Workspace carve (256B aligned): bf16 latents + f32 agg + packed weights.
  size_t off = 0;
  auto carve = [&](size_t bytes) -> void* {
    void* r = (char*)d_ws + off;
    off = (off + bytes + 255) & ~(size_t)255;
    return r;
  };
  bf16_t* x_enc = (bf16_t*)carve((size_t)kNodes * kLatent * 2);
  bf16_t* x_cur = (bf16_t*)carve((size_t)kNodes * kLatent * 2);
  bf16_t* e_enc = (bf16_t*)carve((size_t)kEdges * kLatent * 2);
  bf16_t* e_cur = (bf16_t*)carve((size_t)kEdges * kLatent * 2);
  float*  agg   = (float*)carve((size_t)kNodes * kLatent * 4);
  bf16_t* pk    = (bf16_t*)carve((size_t)1587200 * 2);

  // Packed weight offsets (elements).
  bf16_t* pk_encn = pk;                        // 8192 + 16384 + 16384
  bf16_t* pk_ence = pk + 40960;                // 4096 + 16384 + 16384
  bf16_t* pk_dec  = pk + 77824;                // 16384 + 16384 + 2048
  bf16_t* pk_pe   = pk + 112640;               // per-step stride 81920
  bf16_t* pk_pn   = pk + 931840;               // per-step stride 65536

  PackArgs PA; int nd = 0;
  auto add = [&](const float* s, bf16_t* d, int Kact, int KT, int Nact, int NT) {
    PA.d[nd++] = PackDesc{s, d, Kact, KT, Nact, NT};
  };
  add(encn[0], pk_encn,          kNodeIn, 2, 128, 8);
  add(encn[2], pk_encn + 8192,   128, 4, 128, 8);
  add(encn[4], pk_encn + 24576,  128, 4, 128, 8);
  add(ence[0], pk_ence,          kEdgeIn, 1, 128, 8);
  add(ence[2], pk_ence + 4096,   128, 4, 128, 8);
  add(ence[4], pk_ence + 20480,  128, 4, 128, 8);
  add(dec[0],  pk_dec,           128, 4, 128, 8);
  add(dec[2],  pk_dec + 16384,   128, 4, 128, 8);
  add(dec[4],  pk_dec + 32768,   128, 4, 3, 1);
  for (int s = 0; s < kSteps; ++s) {
    add(pe[0] + (size_t)s * 384 * 128, pk_pe + (size_t)s * 81920,         384, 12, 128, 8);
    add(pe[2] + (size_t)s * 16384,     pk_pe + (size_t)s * 81920 + 49152, 128, 4, 128, 8);
    add(pe[4] + (size_t)s * 16384,     pk_pe + (size_t)s * 81920 + 65536, 128, 4, 128, 8);
  }
  for (int s = 0; s < kSteps; ++s) {
    add(pn[0] + (size_t)s * 256 * 128, pk_pn + (size_t)s * 65536,         256, 8, 128, 8);
    add(pn[2] + (size_t)s * 16384,     pk_pn + (size_t)s * 65536 + 32768, 128, 4, 128, 8);
    add(pn[4] + (size_t)s * 16384,     pk_pn + (size_t)s * 65536 + 49152, 128, 4, 128, 8);
  }
  pack_weights_kernel<<<dim3(32, nd), 256, 0, stream>>>(PA);
  hipMemsetAsync(agg, 0, (size_t)kNodes * kLatent * 4, stream);

  const int nodeBlocks = (kNodes / 16 + 3) / 4;   // 313
  const int edgeBlocks = (kEdges / 16 + 3) / 4;   // 5000

  {  // node encoder
    MlpArgs a{}; a.a_f32 = x; a.rows = kNodes;
    a.W0 = pk_encn; a.W1 = pk_encn + 8192; a.W2 = pk_encn + 24576;
    a.b0 = encn[1]; a.b1 = encn[3]; a.b2 = encn[5];
    a.out_enc = x_enc; a.out_cur = x_cur;
    mlp3_kernel<M_ENC_NODE, 2><<<nodeBlocks, 128, 0, stream>>>(a);
  }
  {  // edge encoder
    MlpArgs a{}; a.a_f32 = eattr; a.rows = kEdges;
    a.W0 = pk_ence; a.W1 = pk_ence + 4096; a.W2 = pk_ence + 20480;
    a.b0 = ence[1]; a.b1 = ence[3]; a.b2 = ence[5];
    a.out_enc = e_enc; a.out_cur = e_cur;
    mlp3_kernel<M_ENC_EDGE, 1><<<edgeBlocks, 128, 0, stream>>>(a);
  }
  for (int s = 0; s < kSteps; ++s) {
    {  // edge block: MLP + LN, residual into e_cur, scatter-add into agg
      MlpArgs a{}; a.rows = kEdges;
      a.e_in = e_cur; a.x_in = x_cur; a.recv = recv; a.send = send;
      a.W0 = pk_pe + (size_t)s * 81920;
      a.W1 = pk_pe + (size_t)s * 81920 + 49152;
      a.W2 = pk_pe + (size_t)s * 81920 + 65536;
      a.b0 = pe[1] + (size_t)s * 128; a.b1 = pe[3] + (size_t)s * 128;
      a.b2 = pe[5] + (size_t)s * 128;
      a.res = e_enc; a.out_cur = e_cur; a.agg_rmw = agg;
      mlp3_kernel<M_PROC_EDGE, 12><<<edgeBlocks, 128, 0, stream>>>(a);
    }
    {  // node block: MLP + LN, residual into x_cur, re-zero agg
      MlpArgs a{}; a.rows = kNodes;
      a.x_in = x_cur; a.agg_in = agg;
      a.W0 = pk_pn + (size_t)s * 65536;
      a.W1 = pk_pn + (size_t)s * 65536 + 32768;
      a.W2 = pk_pn + (size_t)s * 65536 + 49152;
      a.b0 = pn[1] + (size_t)s * 128; a.b1 = pn[3] + (size_t)s * 128;
      a.b2 = pn[5] + (size_t)s * 128;
      a.res = x_enc; a.out_cur = x_cur; a.agg_rmw = agg;
      mlp3_kernel<M_PROC_NODE, 8><<<nodeBlocks, 128, 0, stream>>>(a);
    }
  }
  {  // decoder
    MlpArgs a{}; a.rows = kNodes; a.x_in = x_cur; a.nb2 = 3;
    a.W0 = pk_dec; a.W1 = pk_dec + 16384; a.W2 = pk_dec + 32768;
    a.b0 = dec[1]; a.b1 = dec[3]; a.b2 = dec[5];
    a.dec_out = (float*)d_out;
    mlp3_kernel<M_DEC, 4><<<nodeBlocks, 128, 0, stream>>>(a);
  }
}
